// AdaptiveFourierNeuralOperator_30855045054583
// MI455X (gfx1250) — matrix operand order repaired and checked by reading, not compiled, not run on hardware
//
#include <hip/hip_runtime.h>
#include <hip/hip_bf16.h>
#include <math.h>

// ---------------------------------------------------------------------------
// AFNO for MI455X (gfx1250, wave32, WMMA)
//   B=16, H=32, W=64, N=2048, C=768, NB=4, BS=192, Wf=33, T=16*32*33=16896
// ---------------------------------------------------------------------------

typedef __attribute__((ext_vector_type(16))) __bf16 v16bf;
typedef __attribute__((ext_vector_type(8)))  float  v8f;
typedef __attribute__((ext_vector_type(2)))  float  v2f;
typedef __attribute__((ext_vector_type(4)))  int    v4i;

#define PI_F 3.14159265358979323846f

// ---- async global->LDS copy (gfx1250) with compile-time fallback ----------
#if __has_builtin(__builtin_amdgcn_global_load_async_to_lds_b128) && \
    __has_builtin(__builtin_amdgcn_s_wait_asynccnt)
#define USE_ASYNC_COPY 1
#else
#define USE_ASYNC_COPY 0
#endif

#if USE_ASYNC_COPY
typedef __attribute__((address_space(1))) v4i as1_v4i;
typedef __attribute__((address_space(3))) v4i as3_v4i;
__device__ __forceinline__ void async_copy16(const float* g, float* l) {
  __builtin_amdgcn_global_load_async_to_lds_b128((as1_v4i*)g, (as3_v4i*)l, 0, 0);
}
__device__ __forceinline__ void async_wait0() {
  __builtin_amdgcn_s_wait_asynccnt(0);
}
#endif

__device__ __forceinline__ unsigned short f2bf_hi(float f) {
  unsigned u = __float_as_uint(f);
  unsigned r = u + 0x7FFFu + ((u >> 16) & 1u);   // round-to-nearest-even
  return (unsigned short)(r >> 16);
}
__device__ __forceinline__ float bfbits2f(unsigned short s) {
  return __uint_as_float(((unsigned)s) << 16);
}

union FragBF { uint4 q[2]; v16bf v; };

// ---------------------------------------------------------------------------
// Kernel 1: bias path  out[m,n] = sum_k x[m,k]*W[n,k] + bias[n]
//   M=32768, N=768, K=768.  Split-bf16 (hi+lo) 3-product WMMA ~= fp32.
//   Block tile 128x64, 8 waves (4x2), each wave 32x32 (2x2 WMMA tiles).
//   LDS layout is 128-bit-fragment friendly: A [m][k], B transposed [n][k],
//   row stride 40 shorts (80B, 16B aligned) -> each fragment = 2x ds_load_b128.
// ---------------------------------------------------------------------------
__global__ __launch_bounds__(256) void bias_gemm_kernel(
    const float* __restrict__ X, const float* __restrict__ W,
    const float* __restrict__ bias, float* __restrict__ out) {
  const int m0 = blockIdx.x * 128;
  const int n0 = blockIdx.y * 64;
  const int t  = threadIdx.x;
  const int lane = t & 31, wave = t >> 5;
  const int waveM = wave & 3, waveN = wave >> 2;
  const int hl = lane >> 4, l15 = lane & 15;

  __shared__ __align__(16) unsigned short Ahi[128][40], Alo[128][40];
  __shared__ __align__(16) unsigned short Bhi[64][40],  Blo[64][40];  // [n][k]

  v8f zero = {};
  v8f acc[2][2] = {{zero, zero}, {zero, zero}};

  for (int kc = 0; kc < 768; kc += 32) {
    if (kc + 32 < 768) {
      __builtin_prefetch(&X[(m0 + (t >> 1)) * 768 + kc + 32], 0, 1);
      __builtin_prefetch(&W[(n0 + (t & 63)) * 768 + kc + 32], 0, 1);
    }
    // A tile 128x32 (fp32 -> hi/lo bf16), float4 global loads
#pragma unroll
    for (int i = 0; i < 4; ++i) {
      int off = (i * 256 + t) * 4;
      int r = off >> 5, c = off & 31;
      const float4 v4 = *(const float4*)&X[(m0 + r) * 768 + kc + c];
      float vv[4] = {v4.x, v4.y, v4.z, v4.w};
#pragma unroll
      for (int j = 0; j < 4; ++j) {
        unsigned short h = f2bf_hi(vv[j]);
        Ahi[r][c + j] = h;
        Alo[r][c + j] = f2bf_hi(vv[j] - bfbits2f(h));
      }
    }
    // B tile: B[k][n] = W[(n0+n)*768 + kc+k], stored transposed as [n][k]
#pragma unroll
    for (int i = 0; i < 2; ++i) {
      int off = (i * 256 + t) * 4;
      int k = off & 31, n = off >> 5;
      const float4 v4 = *(const float4*)&W[(n0 + n) * 768 + kc + k];
      float vv[4] = {v4.x, v4.y, v4.z, v4.w};
#pragma unroll
      for (int j = 0; j < 4; ++j) {
        unsigned short h = f2bf_hi(vv[j]);
        Bhi[n][k + j] = h;
        Blo[n][k + j] = f2bf_hi(vv[j] - bfbits2f(h));
      }
    }
    __syncthreads();

    // Fragments per ISA 7.12.2:
    //   A(16x32): elem j<8 -> k=8*half+j ; j>=8 -> k=16+8*half+(j-8)
    //   B(32x16): same k order as A
    FragBF fah[2], fal[2], fbh[2], fbl[2];
#pragma unroll
    for (int tm = 0; tm < 2; ++tm) {
      int m = waveM * 32 + tm * 16 + l15;
      const uint4* ph = (const uint4*)__builtin_assume_aligned(&Ahi[m][0], 16);
      const uint4* pl = (const uint4*)__builtin_assume_aligned(&Alo[m][0], 16);
      fah[tm].q[0] = ph[hl];     fah[tm].q[1] = ph[2 + hl];
      fal[tm].q[0] = pl[hl];     fal[tm].q[1] = pl[2 + hl];
    }
#pragma unroll
    for (int tn = 0; tn < 2; ++tn) {
      int n = waveN * 32 + tn * 16 + l15;
      const uint4* ph = (const uint4*)__builtin_assume_aligned(&Bhi[n][0], 16);
      const uint4* pl = (const uint4*)__builtin_assume_aligned(&Blo[n][0], 16);
      fbh[tn].q[0] = ph[hl]; fbh[tn].q[1] = ph[2 + hl];
      fbl[tn].q[0] = pl[hl]; fbl[tn].q[1] = pl[2 + hl];
    }
#pragma unroll
    for (int tm = 0; tm < 2; ++tm)
#pragma unroll
      for (int tn = 0; tn < 2; ++tn) {
        acc[tm][tn] = __builtin_amdgcn_wmma_f32_16x16x32_bf16(
            false, fah[tm].v, false, fbh[tn].v, (short)0, acc[tm][tn], false, false);
        acc[tm][tn] = __builtin_amdgcn_wmma_f32_16x16x32_bf16(
            false, fah[tm].v, false, fbl[tn].v, (short)0, acc[tm][tn], false, false);
        acc[tm][tn] = __builtin_amdgcn_wmma_f32_16x16x32_bf16(
            false, fal[tm].v, false, fbh[tn].v, (short)0, acc[tm][tn], false, false);
      }
    __syncthreads();
  }

#pragma unroll
  for (int tm = 0; tm < 2; ++tm)
#pragma unroll
    for (int tn = 0; tn < 2; ++tn) {
      int col = n0 + waveN * 32 + tn * 16 + l15;
      float bv = bias[col];
#pragma unroll
      for (int i = 0; i < 8; ++i) {
        int row = m0 + waveM * 32 + tm * 16 + i + 8 * hl;  // C-tile: VGPR i -> M = i + 8*half
        out[row * 768 + col] = acc[tm][tn][i] + bv;
      }
    }
}

// ---------------------------------------------------------------------------
// Kernel 2: spectral block GEMM (fp32 WMMA 16x16x4)
//   out[t, nb*192+n] = act( sum_d P[t,d]*Wp[nb,d,n] + sgn * Q[t,d]*Wq[nb,d,n] + bias )
//   M=16896 tokens, K=192, N=192 per freq-block; grid.z = nb.
// ---------------------------------------------------------------------------
__global__ __launch_bounds__(256) void spec_gemm_kernel(
    const float* __restrict__ P, const float* __restrict__ Q,
    const float* __restrict__ Wp, const float* __restrict__ Wq,
    const float* __restrict__ bias, float* __restrict__ out,
    float sgn, int relu) {
  const int t0 = blockIdx.x * 128;
  const int nt = blockIdx.y * 64;
  const int nb = blockIdx.z;
  const int cbase = nb * 192;
  const int t = threadIdx.x, lane = t & 31, wave = t >> 5;
  const int waveM = wave & 3, waveN = wave >> 2;
  const int hl = lane >> 4, l15 = lane & 15;

  __shared__ __align__(16) float Pa[128][20], Qa[128][20];
  __shared__ __align__(16) float Wpt[64][18], Wqt[64][18];  // transposed [n][k]

  v8f zero = {};
  v8f acc[2][2] = {{zero, zero}, {zero, zero}};

  const float* wp_g = Wp + nb * 192 * 192;
  const float* wq_g = Wq + nb * 192 * 192;

  for (int kc = 0; kc < 192; kc += 16) {
    // A tiles 128x16, float4 global loads, float4 LDS stores
#pragma unroll
    for (int i = 0; i < 2; ++i) {
      int off = (i * 256 + t) * 4;
      int r = off >> 4, k = off & 15;
      *(float4*)&Pa[r][k] = *(const float4*)&P[(t0 + r) * 768 + cbase + kc + k];
      *(float4*)&Qa[r][k] = *(const float4*)&Q[(t0 + r) * 768 + cbase + kc + k];
    }
    // W tiles 16x64 -> transposed [n][k] (fold sgn into Wq; f32 WMMA has no A/B neg)
    {
      int off = t * 4;
      int kd = off >> 6, n = off & 63;
      const float4 vp = *(const float4*)&wp_g[(kc + kd) * 192 + nt + n];
      const float4 vq = *(const float4*)&wq_g[(kc + kd) * 192 + nt + n];
      float pp[4] = {vp.x, vp.y, vp.z, vp.w};
      float qq[4] = {vq.x, vq.y, vq.z, vq.w};
#pragma unroll
      for (int j = 0; j < 4; ++j) {
        Wpt[n + j][kd] = pp[j];
        Wqt[n + j][kd] = sgn * qq[j];
      }
    }
    __syncthreads();

#pragma unroll
    for (int ks = 0; ks < 4; ++ks) {
      int kb = ks * 4;
      // f32 A 16x4: lane m=lane&15, VGPR j -> K = 2*half + j; B 4x16 mirrors.
      v2f ap[2], aq[2], fp[2], fq[2];
#pragma unroll
      for (int tm = 0; tm < 2; ++tm) {
        int m = waveM * 32 + tm * 16 + l15;
        ap[tm] = *(const v2f*)&Pa[m][kb + 2 * hl];
        aq[tm] = *(const v2f*)&Qa[m][kb + 2 * hl];
      }
#pragma unroll
      for (int tn = 0; tn < 2; ++tn) {
        int n = waveN * 32 + tn * 16 + l15;
        fp[tn] = *(const v2f*)&Wpt[n][kb + 2 * hl];
        fq[tn] = *(const v2f*)&Wqt[n][kb + 2 * hl];
      }
#pragma unroll
      for (int tm = 0; tm < 2; ++tm)
#pragma unroll
        for (int tn = 0; tn < 2; ++tn) {
          acc[tm][tn] = __builtin_amdgcn_wmma_f32_16x16x4_f32(
              false, ap[tm], false, fp[tn], (short)0, acc[tm][tn], false, false);
          acc[tm][tn] = __builtin_amdgcn_wmma_f32_16x16x4_f32(
              false, aq[tm], false, fq[tn], (short)0, acc[tm][tn], false, false);
        }
    }
    __syncthreads();
  }

#pragma unroll
  for (int tm = 0; tm < 2; ++tm)
#pragma unroll
    for (int tn = 0; tn < 2; ++tn) {
      int coll = nt + waveN * 32 + tn * 16 + l15;
      float bv = bias[cbase + coll];
#pragma unroll
      for (int i = 0; i < 8; ++i) {
        int row = t0 + waveM * 32 + tm * 16 + i + 8 * hl;
        float v = acc[tm][tn][i] + bv;
        if (relu) v = fmaxf(v, 0.0f);
        out[row * 768 + cbase + coll] = v;
      }
    }
}

// ---------------------------------------------------------------------------
// Kernel 3: forward rDFT along W (ortho): x(B,H,64,C) -> Yr,Yi(B,H,33,C)
//   One (b,h) row per block (x 4 channel chunks of 192), staged in LDS
//   via async global->LDS b128 copies when available.
// ---------------------------------------------------------------------------
__global__ __launch_bounds__(256) void dftW_kernel(
    const float* __restrict__ x, float* __restrict__ Yr, float* __restrict__ Yi) {
  const int bh = blockIdx.x;            // b*32 + h
  const int cc = blockIdx.y * 192;
  __shared__ __align__(16) float xs[64][192];
  __shared__ float ct[64], st[64];
  const int t = threadIdx.x;
  if (t < 64) {
    float ang = (2.0f * PI_F / 64.0f) * (float)t;
    ct[t] = cosf(ang);
    st[t] = sinf(ang);
  }
#if USE_ASYNC_COPY
#pragma unroll
  for (int i = 0; i < 12; ++i) {
    int off = (i * 256 + t) * 4;
    int w = off / 192, c = off % 192;
    async_copy16(&x[(bh * 64 + w) * 768 + cc + c], &xs[w][c]);
  }
  async_wait0();
#else
#pragma unroll
  for (int i = 0; i < 48; ++i) {
    int lin = i * 256 + t;
    int w = lin / 192, c = lin % 192;
    xs[w][c] = x[(bh * 64 + w) * 768 + cc + c];
  }
#endif
  __syncthreads();
  for (int o = t; o < 33 * 192; o += 256) {
    int wp = o / 192, c = o % 192;
    float sr = 0.f, si = 0.f;
    for (int w = 0; w < 64; ++w) {
      int idx = (w * wp) & 63;
      float v = xs[w][c];
      sr += v * ct[idx];
      si -= v * st[idx];
    }
    int oi = (bh * 33 + wp) * 768 + cc + c;
    Yr[oi] = 0.125f * sr;     // 1/sqrt(64)
    Yi[oi] = 0.125f * si;
  }
}

// ---------------------------------------------------------------------------
// Kernel 4: complex DFT along H (sgn=-1 forward, +1 inverse), ortho 1/sqrt(32)
//   (Ar,Ai)(B,32,33,C) -> (Br,Bi)(B,32,33,C); one (b,w') per block.
// ---------------------------------------------------------------------------
__global__ __launch_bounds__(256) void dftH_kernel(
    const float* __restrict__ Ar, const float* __restrict__ Ai,
    float* __restrict__ Br, float* __restrict__ Bi, float sgn) {
  const int b = blockIdx.x / 33, wp = blockIdx.x % 33;
  const int cc = blockIdx.y * 96;
  __shared__ __align__(16) float yr[32][96], yi[32][96];
  __shared__ float ct[32], st[32];
  const int t = threadIdx.x;
  if (t < 32) {
    float ang = (2.0f * PI_F / 32.0f) * (float)t;
    ct[t] = cosf(ang);
    st[t] = sinf(ang);
  }
#if USE_ASYNC_COPY
#pragma unroll
  for (int i = 0; i < 3; ++i) {
    int off = (i * 256 + t) * 4;
    int h = off / 96, c = off % 96;
    int gi = ((b * 32 + h) * 33 + wp) * 768 + cc + c;
    async_copy16(&Ar[gi], &yr[h][c]);
    async_copy16(&Ai[gi], &yi[h][c]);
  }
  async_wait0();
#else
#pragma unroll
  for (int i = 0; i < 12; ++i) {
    int lin = i * 256 + t;
    int h = lin / 96, c = lin % 96;
    int gi = ((b * 32 + h) * 33 + wp) * 768 + cc + c;
    yr[h][c] = Ar[gi];
    yi[h][c] = Ai[gi];
  }
#endif
  __syncthreads();
  for (int o = t; o < 32 * 96; o += 256) {
    int hp = o / 96, c = o % 96;
    float zr = 0.f, zi = 0.f;
    for (int h = 0; h < 32; ++h) {
      int idx = (h * hp) & 31;
      float r = yr[h][c], im = yi[h][c];
      zr += r * ct[idx] - sgn * im * st[idx];
      zi += im * ct[idx] + sgn * r * st[idx];
    }
    int go = ((b * 32 + hp) * 33 + wp) * 768 + cc + c;
    Br[go] = 0.17677669529663688f * zr;   // 1/sqrt(32)
    Bi[go] = 0.17677669529663688f * zi;
  }
}

// ---------------------------------------------------------------------------
// Kernel 5: inverse rDFT along W (Hermitian) + accumulate onto bias in `out`
// ---------------------------------------------------------------------------
__global__ __launch_bounds__(256) void irfftW_add_kernel(
    const float* __restrict__ Sr, const float* __restrict__ Si,
    float* __restrict__ out) {
  const int bh = blockIdx.x;
  const int cc = blockIdx.y * 192;
  __shared__ __align__(16) float sr[33][192], si[33][192];
  __shared__ float ct[64], st[64];
  const int t = threadIdx.x;
  if (t < 64) {
    float ang = (2.0f * PI_F / 64.0f) * (float)t;
    ct[t] = cosf(ang);
    st[t] = sinf(ang);
  }
#if USE_ASYNC_COPY
  for (int lin = t; lin < 1584; lin += 256) {   // 33*192/4 float4 copies
    int off = lin * 4;
    int k = off / 192, c = off % 192;
    int gi = (bh * 33 + k) * 768 + cc + c;
    async_copy16(&Sr[gi], &sr[k][c]);
    async_copy16(&Si[gi], &si[k][c]);
  }
  async_wait0();
#else
  for (int lin = t; lin < 33 * 192; lin += 256) {
    int k = lin / 192, c = lin % 192;
    int gi = (bh * 33 + k) * 768 + cc + c;
    sr[k][c] = Sr[gi];
    si[k][c] = Si[gi];
  }
#endif
  __syncthreads();
  for (int o = t; o < 64 * 192; o += 256) {
    int w = o / 192, c = o % 192;
    float acc = sr[0][c];                       // k=0 term (real only)
    for (int k = 1; k < 32; ++k) {
      int idx = (w * k) & 63;
      acc += 2.0f * (sr[k][c] * ct[idx] - si[k][c] * st[idx]);
    }
    acc += (w & 1) ? -sr[32][c] : sr[32][c];    // k=32: cos(pi*w)=(-1)^w
    int go = (bh * 64 + w) * 768 + cc + c;
    out[go] += 0.125f * acc;                    // 1/sqrt(64)
  }
}

// ---------------------------------------------------------------------------
extern "C" void kernel_launch(void* const* d_in, const int* in_sizes, int n_in,
                              void* d_out, int out_size, void* d_ws, size_t ws_size,
                              hipStream_t stream) {
  const float* x      = (const float*)d_in[0];
  const float* w1     = (const float*)d_in[1];
  const float* b1     = (const float*)d_in[2];
  const float* w2     = (const float*)d_in[3];
  const float* b2     = (const float*)d_in[4];
  const float* bias_w = (const float*)d_in[5];
  const float* bias_b = (const float*)d_in[6];
  float* out = (float*)d_out;

  const size_t SPEC = (size_t)16 * 32 * 33 * 768;   // 12,976,128 floats
  float* buf = (float*)d_ws;
  float* Yr = buf + 0 * SPEC;
  float* Yi = buf + 1 * SPEC;
  float* Zr = buf + 2 * SPEC;
  float* Zi = buf + 3 * SPEC;
  float* U0 = buf + 4 * SPEC;   // R1
  float* U1 = buf + 5 * SPEC;   // I1

  // 1) bias path: out = x @ bias_w^T + bias_b   (split-bf16 WMMA GEMM)
  hipLaunchKernelGGL(bias_gemm_kernel, dim3(256, 12), dim3(256), 0, stream,
                     x, bias_w, bias_b, out);

  // 2) rfft2 (ortho) as two DFT stages
  hipLaunchKernelGGL(dftW_kernel, dim3(512, 4), dim3(256), 0, stream, x, Yr, Yi);
  hipLaunchKernelGGL(dftH_kernel, dim3(528, 8), dim3(256), 0, stream,
                     Yr, Yi, Zr, Zi, -1.0f);

  // 3) spectral block-diagonal complex MLP (fp32 WMMA GEMMs)
  const float* w1r = w1;  const float* w1i = w1 + 4 * 192 * 192;
  const float* w2r = w2;  const float* w2i = w2 + 4 * 192 * 192;
  const float* b1r = b1;  const float* b1i = b1 + 4 * 192;
  const float* b2r = b2;  const float* b2i = b2 + 4 * 192;
  dim3 sg(132, 3, 4);
  // R1 = relu(Zr*w1r - Zi*w1i + b1r)
  hipLaunchKernelGGL(spec_gemm_kernel, sg, dim3(256), 0, stream,
                     Zr, Zi, w1r, w1i, b1r, U0, -1.0f, 1);
  // I1 = relu(Zr*w1i + Zi*w1r + b1i)
  hipLaunchKernelGGL(spec_gemm_kernel, sg, dim3(256), 0, stream,
                     Zr, Zi, w1i, w1r, b1i, U1, 1.0f, 1);
  // R2 = R1*w2r - I1*w2i + b2r   (-> Yr, now free)
  hipLaunchKernelGGL(spec_gemm_kernel, sg, dim3(256), 0, stream,
                     U0, U1, w2r, w2i, b2r, Yr, -1.0f, 0);
  // I2 = R2*w2i + I1*w2r + b2i   (uses updated R2, matching reference; -> Yi)
  hipLaunchKernelGGL(spec_gemm_kernel, sg, dim3(256), 0, stream,
                     Yr, U1, w2i, w2r, b2i, Yi, 1.0f, 0);

  // 4) irfft2 (ortho): inverse H DFT, then inverse W rDFT accumulated onto bias
  hipLaunchKernelGGL(dftH_kernel, dim3(528, 8), dim3(256), 0, stream,
                     Yr, Yi, Zr, Zi, 1.0f);
  hipLaunchKernelGGL(irfftW_add_kernel, dim3(512, 4), dim3(256), 0, stream,
                     Zr, Zi, out);

  (void)in_sizes; (void)n_in; (void)out_size; (void)ws_size;
}
